// CapsuleNetwork_55310588838168
// MI455X (gfx1250) — matrix operand order repaired and checked by reading, not compiled, not run on hardware
//
#include <hip/hip_runtime.h>
#include <hip/hip_bf16.h>

typedef _Float16 v16h __attribute__((ext_vector_type(16)));
typedef _Float16 v8h  __attribute__((ext_vector_type(8)));
typedef float    v8f  __attribute__((ext_vector_type(8)));
typedef unsigned int uint32x4 __attribute__((ext_vector_type(4)));
typedef int          int32x8  __attribute__((ext_vector_type(8)));
typedef int          int32x4  __attribute__((ext_vector_type(4)));

#define KTOT  (81 * 256)        // 20736  (conv2 GEMM K)
#define MTOT  (128 * 12 * 12)   // 18432  (conv2 GEMM M)
#define PCAPS 4608
#define ROUT_SMEM (36864 * 2 + 2560 * 4 + 320 * 4)   // 85248 B dynamic LDS

// TDM: builtin exists on both toolchains but with different arity.
// ROCm 7.2 (clang-22): 5 args.  amdgpu-toolchain (clang-23, therock headers,
// which ship amd_gfx1250_TDM.h): 6 args (extra int32x8 group before cpol).
// Use header presence only to pick the ARITY; use __has_builtin to gate use.
#if defined(__HIP_DEVICE_COMPILE__) && __has_builtin(__builtin_amdgcn_tensor_load_to_lds)
#define USE_TDM 1
#else
#define USE_TDM 0
#endif
#if __has_include(<hip/amd_detail/amd_gfx1250_TDM.h>)
#define TDM_6ARG 1
#else
#define TDM_6ARG 0
#endif

// ---------------------------------------------------------------------------
// conv1: 9x9, 1->256, stride 1, VALID, ReLU.  One block per (b,oy,ox), 256 oc.
// ---------------------------------------------------------------------------
__global__ void conv1_relu_f16(const float* __restrict__ x,
                               const float* __restrict__ w1,
                               const float* __restrict__ b1,
                               _Float16* __restrict__ h) {
  __shared__ float xs[81];
  const int blk = blockIdx.x;          // b*1024 + oy*32 + ox
  const int b  = blk >> 10;
  const int oy = (blk >> 5) & 31;
  const int ox = blk & 31;
  const int t  = threadIdx.x;
  if (t < 81) {
    int ky = t / 9, kx = t % 9;
    xs[t] = x[(b * 40 + oy + ky) * 40 + (ox + kx)];
  }
  __syncthreads();
  const int oc = t;
  float acc = b1[oc];
  #pragma unroll 9
  for (int tap = 0; tap < 81; ++tap)
    acc = fmaf(xs[tap], w1[tap * 256 + oc], acc);
  acc = fmaxf(acc, 0.0f);
  h[(size_t)blk * 256 + oc] = (_Float16)acc;
}

// ---------------------------------------------------------------------------
// conv2 weights (9,9,256,256) HWIO -> f16 transposed [oc][k]
// ---------------------------------------------------------------------------
__global__ void conv2_w_to_f16_t(const float* __restrict__ w2,
                                 _Float16* __restrict__ w2t) {
  const int idx = blockIdx.x * 256 + threadIdx.x;   // < 20736*256
  const int oc = idx & 255;
  const int k  = idx >> 8;
  w2t[(size_t)oc * KTOT + k] = (_Float16)w2[(size_t)k * 256 + oc];
}

// ---------------------------------------------------------------------------
// conv2 implicit GEMM, WMMA f16->f32.  M=18432, N=256, K=20736.
// Block tile 128(M) x 64(N), BK=32, 8 wave32 waves in a 4x2 grid; each wave
// owns a 32x32 tile = 4 accumulators = 4 WMMAs per K-step with A/B reuse.
// LDS double-buffered -> single barrier per K-step.  B tile fetched by the
// Tensor Data Mover (2D D#), A tile by coalesced 16B loads.
// ---------------------------------------------------------------------------
__global__ void conv2_wmma(const _Float16* __restrict__ h,
                           const _Float16* __restrict__ w2t,
                           const float* __restrict__ b2,
                           float* __restrict__ pc) {
  __shared__ __align__(64) _Float16 As[2][128][32];
  __shared__ __align__(64) _Float16 Bs[2][64][32];   // [n][k]

  const int t    = threadIdx.x;
  const int lane = t & 31;
  const int warp = t >> 5;
  const int bm0  = blockIdx.x * 128;
  const int nb0  = blockIdx.y * 64;
  const int m0   = (warp & 3) * 32;     // wave M sub-tile (32 rows)
  const int n0   = (warp >> 2) * 32;    // wave N sub-tile (32 cols)

  // A-tile cooperative load: 128 rows x 32 halves, 16 halves/thread.
  const int arow = t >> 1;
  const int acol = (t & 1) * 16;
  const int m_g  = bm0 + arow;          // < 18432, always in-bounds
  const int bidx = m_g / 144;
  const int rem  = m_g % 144;
  const int oy   = rem / 12;
  const int ox   = rem % 12;

#if !USE_TDM
  // B-tile cooperative load: 64 rows(n) x 32 halves(k), 8 halves/thread.
  const int brow = t >> 2;
  const int bcol = (t & 3) * 8;
#endif

  // WMMA fragment lane mapping (CDNA5 ISA layouts).
  const int fhalf = lane >> 4;          // 0/1
  const int fidx  = lane & 15;          // A: M row;  B: N col
  const int ak0   = fhalf * 8;          // A lane K origin {0,8}, plus +16 chunk

  auto load_tile = [&](int kk, int nb) {
    const int tap = kk >> 8;
    const int icb = kk & 255;
    const int ky = tap / 9, kx = tap % 9;
    const int iy = oy * 2 + ky, ix = ox * 2 + kx;
    const _Float16* aptr =
        h + ((size_t)((bidx * 32 + iy) * 32 + ix)) * 256 + icb + acol;
    *(v8h*)&As[nb][arow][acol]     = *(const v8h*)aptr;
    *(v8h*)&As[nb][arow][acol + 8] = *(const v8h*)(aptr + 8);
    __builtin_prefetch((const void*)(aptr + 32), 0, 1);
#if USE_TDM
    if (warp == 0) {
      // 2D TDM descriptor: tile 64(n) x 32(k) halves from w2t[nb0..][kk..]
      unsigned long long ga =
          (unsigned long long)(uintptr_t)(w2t + (size_t)nb0 * KTOT + kk);
      unsigned lds_b = (unsigned)(uintptr_t)&Bs[nb][0][0];
      uint32x4 g0 = {};
      g0[0] = 1u;                                   // count=1, no gather
      g0[1] = lds_b;                                // lds_addr
      g0[2] = (unsigned)(ga & 0xffffffffu);         // global_addr[31:0]
      g0[3] = (unsigned)(ga >> 32) | 0x80000000u;   // global_addr[56:32]|type=2
      int32x8 g1 = {};
      g1[0] = 0x00010000;            // workgroup_mask=0, data_size=2B
      g1[1] = (int)(20736u << 16);   // tensor_dim0[15:0] at bits 63:48
      g1[2] = (int)(256u << 16);     // tensor_dim0 hi=0 | tensor_dim1[15:0]
      g1[3] = (int)(32u << 16);      // tensor_dim1 hi=0 | tile_dim0=32
      g1[4] = 64;                    // tile_dim1=64, tile_dim2=0
      g1[5] = 20736;                 // tensor_dim0_stride[31:0]
      g1[6] = 0;
      g1[7] = 0;
      int32x4 z4 = {};
#if TDM_6ARG
      int32x8 z8 = {};
      __builtin_amdgcn_tensor_load_to_lds(g0, g1, z4, z4, z8, 0);
#else
      __builtin_amdgcn_tensor_load_to_lds(g0, g1, z4, z4, 0);
#endif
    }
#else
    const _Float16* bp = w2t + (size_t)(nb0 + brow) * KTOT + kk + bcol;
    *(v8h*)&Bs[nb][brow][bcol] = *(const v8h*)bp;
#endif
  };

  // Prologue: stage tile 0 into buffer 0.
  load_tile(0, 0);
#if USE_TDM
  __builtin_amdgcn_s_wait_tensorcnt(0);
#endif
  __syncthreads();

  v8f c00 = {}, c01 = {}, c10 = {}, c11 = {};
  int buf = 0;
  for (int kk = 0; kk < KTOT; kk += 32) {
    // Fragments from current buffer.
    const _Float16* ar0 = &As[buf][m0 + fidx][0];
    const _Float16* ar1 = &As[buf][m0 + 16 + fidx][0];
    v8h a0l = *(const v8h*)(ar0 + ak0);
    v8h a0h = *(const v8h*)(ar0 + ak0 + 16);
    v8h a1l = *(const v8h*)(ar1 + ak0);
    v8h a1h = *(const v8h*)(ar1 + ak0 + 16);
    v16h a0 = __builtin_shufflevector(a0l, a0h, 0,1,2,3,4,5,6,7,8,9,10,11,12,13,14,15);
    v16h a1 = __builtin_shufflevector(a1l, a1h, 0,1,2,3,4,5,6,7,8,9,10,11,12,13,14,15);
    v16h b0 = __builtin_shufflevector(*(const v8h*)&Bs[buf][n0 + fidx][ak0], *(const v8h*)&Bs[buf][n0 + fidx][ak0 + 16], 0,1,2,3,4,5,6,7,8,9,10,11,12,13,14,15);
    v16h b1 = __builtin_shufflevector(*(const v8h*)&Bs[buf][n0 + 16 + fidx][ak0], *(const v8h*)&Bs[buf][n0 + 16 + fidx][ak0 + 16], 0,1,2,3,4,5,6,7,8,9,10,11,12,13,14,15);

    c00 = __builtin_amdgcn_wmma_f32_16x16x32_f16(false, a0, false, b0, (short)0, c00, false, false);
    c01 = __builtin_amdgcn_wmma_f32_16x16x32_f16(false, a0, false, b1, (short)0, c01, false, false);
    c10 = __builtin_amdgcn_wmma_f32_16x16x32_f16(false, a1, false, b0, (short)0, c10, false, false);
    c11 = __builtin_amdgcn_wmma_f32_16x16x32_f16(false, a1, false, b1, (short)0, c11, false, false);

    if (kk + 32 < KTOT) {
      load_tile(kk + 32, buf ^ 1);     // prefetch next tile into other buffer
#if USE_TDM
      __builtin_amdgcn_s_wait_tensorcnt(0);
#endif
    }
    __syncthreads();                   // single barrier per K-step
    buf ^= 1;
  }

  // Store D tiles with bias: lane fidx = N col, VGPR r -> M = r + 8*fhalf.
  const int oc0 = nb0 + n0 + fidx;
  const int oc1 = oc0 + 16;
  const float bb0 = b2[oc0];
  const float bb1 = b2[oc1];
  const int mb0 = bm0 + m0 + 8 * fhalf;
  const int mb1 = mb0 + 16;
  #pragma unroll
  for (int r = 0; r < 8; ++r) {
    pc[(size_t)(mb0 + r) * 256 + oc0] = c00[r] + bb0;
    pc[(size_t)(mb0 + r) * 256 + oc1] = c01[r] + bb1;
    pc[(size_t)(mb1 + r) * 256 + oc0] = c10[r] + bb0;
    pc[(size_t)(mb1 + r) * 256 + oc1] = c11[r] + bb1;
  }
}

// ---------------------------------------------------------------------------
// Fused dynamic routing: one block per batch element, u[b] in 74KB dynamic LDS,
// u_hat recomputed on the fly (K=8), 3 iterations, logits in global scratch
// with strict same-thread ownership (no fences needed, never read-before-write).
// ---------------------------------------------------------------------------
__global__ void routing_kernel(const float* __restrict__ pc,
                               const float* __restrict__ w0,   // (4608,10,16,8)
                               const int* __restrict__ y,
                               float* __restrict__ bcoef,      // (B,4608,10)
                               float* __restrict__ v_out,      // (B,10,16)
                               float* __restrict__ vmask) {    // (B,160)
  extern __shared__ char smem[];
  _Float16* u_l = (_Float16*)smem;                       // 36864 halves
  float* c_l = (float*)(smem + 36864 * 2);               // 256*10
  float* s_l = (float*)(smem + 36864 * 2 + 2560 * 4);    // 160
  float* v_l = s_l + 160;                                // 160

  const int b   = blockIdx.x;
  const int tid = threadIdx.x;
  const float* pcb = pc + (size_t)b * 36864;
  for (int i = tid; i < 36864; i += 256) u_l[i] = (_Float16)pcb[i];
  float* bco = bcoef + (size_t)b * (PCAPS * 10);
  __syncthreads();

  const int j_own = tid >> 4, s_own = tid & 15;

  for (int it = 0; it < 3; ++it) {
    // ---- pass 1: c = softmax(b); s[j,s16] = sum_p c * u_hat ----
    float acc = 0.0f;
    for (int chunk = 0; chunk < 18; ++chunk) {
      __syncthreads();
      const int p = chunk * 256 + tid;
      if (it == 0) {
        #pragma unroll
        for (int j = 0; j < 10; ++j) c_l[tid * 10 + j] = 0.1f;   // softmax(0)
      } else {
        float bv[10], mx = -1e30f;
        #pragma unroll
        for (int j = 0; j < 10; ++j) { bv[j] = bco[p * 10 + j]; mx = fmaxf(mx, bv[j]); }
        float ssum = 0.0f;
        #pragma unroll
        for (int j = 0; j < 10; ++j) { bv[j] = __expf(bv[j] - mx); ssum += bv[j]; }
        const float inv = 1.0f / ssum;
        #pragma unroll
        for (int j = 0; j < 10; ++j) c_l[tid * 10 + j] = bv[j] * inv;
      }
      __syncthreads();
      if (tid < 160) {
        for (int i = 0; i < 256; ++i) {
          const int pp = chunk * 256 + i;
          const float* wr = w0 + ((size_t)(pp * 10 + j_own) * 16 + s_own) * 8;
          const v8h uv = *(const v8h*)(u_l + pp * 8);
          float uh = 0.0f;
          #pragma unroll
          for (int vv = 0; vv < 8; ++vv) uh = fmaf(wr[vv], (float)uv[vv], uh);
          acc = fmaf(c_l[i * 10 + j_own], uh, acc);
        }
      }
    }
    if (tid < 160) s_l[tid] = acc;
    __syncthreads();

    // ---- pass 2: squash ----
    if (tid < 10) {
      float sq = 0.0f;
      #pragma unroll
      for (int s = 0; s < 16; ++s) { float vs = s_l[tid * 16 + s]; sq = fmaf(vs, vs, sq); }
      const float scale = sq / ((1.0f + sq) * (sqrtf(sq) + 1e-7f));
      #pragma unroll
      for (int s = 0; s < 16; ++s) v_l[tid * 16 + s] = scale * s_l[tid * 16 + s];
    }
    __syncthreads();

    // ---- pass 3: agreement; b update (skipped on last iter, unused) ----
    if (it < 2) {
      for (int p = tid; p < PCAPS; p += 256) {
        const v8h uv = *(const v8h*)(u_l + p * 8);
        float uf[8];
        #pragma unroll
        for (int vv = 0; vv < 8; ++vv) uf[vv] = (float)uv[vv];
        #pragma unroll 2
        for (int j = 0; j < 10; ++j) {
          const float* wj = w0 + ((size_t)(p * 10 + j) * 16) * 8;   // 128 contiguous
          float agr = 0.0f;
          for (int s = 0; s < 16; ++s) {
            float uh = 0.0f;
            #pragma unroll
            for (int vv = 0; vv < 8; ++vv) uh = fmaf(wj[s * 8 + vv], uf[vv], uh);
            agr = fmaf(uh, v_l[j * 16 + s], agr);
          }
          if (it == 0) bco[p * 10 + j] = agr;
          else         bco[p * 10 + j] += agr;
        }
      }
    }
  }

  if (tid < 160) {
    const float val = v_l[tid];
    v_out[(size_t)b * 160 + tid] = val;
    vmask[(size_t)b * 160 + tid] = (j_own == y[b]) ? val : 0.0f;
  }
}

// ---------------------------------------------------------------------------
// Decoder dense layers: out[b][n] = act(sum_k in[b][k]*W[k][n] + bias[n])
// ---------------------------------------------------------------------------
__global__ void dense_kernel(const float* __restrict__ in,
                             const float* __restrict__ W,
                             const float* __restrict__ bias,
                             float* __restrict__ out,
                             int K, int N, int act) {
  const int n = blockIdx.x * 256 + threadIdx.x;
  const int b = blockIdx.y;
  if (n >= N) return;
  const float* inb = in + (size_t)b * K;
  float acc = bias[n];
  for (int k = 0; k < K; ++k) acc = fmaf(inb[k], W[(size_t)k * N + n], acc);
  acc = (act == 0) ? fmaxf(acc, 0.0f) : 1.0f / (1.0f + __expf(-acc));
  out[(size_t)b * N + n] = acc;
}

// ---------------------------------------------------------------------------
extern "C" void kernel_launch(void* const* d_in, const int* in_sizes, int n_in,
                              void* d_out, int out_size, void* d_ws, size_t ws_size,
                              hipStream_t stream) {
  const float* x   = (const float*)d_in[0];
  const float* w1  = (const float*)d_in[1];
  const float* b1  = (const float*)d_in[2];
  const float* w2  = (const float*)d_in[3];
  const float* b2  = (const float*)d_in[4];
  const float* w0  = (const float*)d_in[5];   // (1,4608,10,16,8)
  const float* d1w = (const float*)d_in[6];
  const float* d1b = (const float*)d_in[7];
  const float* d2w = (const float*)d_in[8];
  const float* d2b = (const float*)d_in[9];
  const float* d3w = (const float*)d_in[10];
  const float* d3b = (const float*)d_in[11];
  const int*   y   = (const int*)d_in[12];
  float* out = (float*)d_out;                 // [0,20480): v ; [20480,...): recon

  char* ws = (char*)d_ws;
  _Float16* h_f16 = (_Float16*)(ws + 0);            //  67,108,864 B
  _Float16* w2t   = (_Float16*)(ws + 67108864);     //  10,616,832 B
  float* pc       = (float*)(ws + 77725696);        //  18,874,368 B
  float* bcoef    = (float*)(ws + 96600064);        //  23,592,960 B
  float* vmask    = (float*)(ws + 120193024);       //      81,920 B
  float* r1       = (float*)(ws + 120274944);       //     262,144 B
  float* r2       = (float*)(ws + 120537088);       //     524,288 B

  conv1_relu_f16<<<128 * 32 * 32, 256, 0, stream>>>(x, w1, b1, h_f16);
  conv2_w_to_f16_t<<<KTOT, 256, 0, stream>>>(w2, w2t);
  conv2_wmma<<<dim3(MTOT / 128, 256 / 64), 256, 0, stream>>>(h_f16, w2t, b2, pc);
  routing_kernel<<<128, 256, ROUT_SMEM, stream>>>(pc, w0, y, bcoef, out, vmask);
  dense_kernel<<<dim3(2, 128), 256, 0, stream>>>(vmask, d1w, d1b, r1, 160, 512, 0);
  dense_kernel<<<dim3(4, 128), 256, 0, stream>>>(r1, d2w, d2b, r2, 512, 1024, 0);
  dense_kernel<<<dim3(7, 128), 256, 0, stream>>>(r2, d3w, d3b, out + 20480, 1024, 1600, 1);
}
